// CLAPP_Sequence_SNN_82686710383112
// MI455X (gfx1250) — compile-verified, matrix operand order checked by reading, not run on hardware
//
#include <hip/hip_runtime.h>
#include <stdint.h>

typedef __attribute__((ext_vector_type(16))) _Float16 v16h;
typedef __attribute__((ext_vector_type(8)))  float    v8f;

#define B_      32
#define T_      4096
#define NIN_    700
#define H1_     1024
#define H2_     1024
#define NOUT_   20

#define K1S     708      // padded LDS row stride (halves) for K=700->704, +4 anti-bank-conflict
#define K2S     1028     // padded LDS row stride for K=1024
#define NK1     22       // 704/32 k-chunks for layer 1
#define NK2     32       // 1024/32 k-chunks for layers 2/out

#define COLS_PER_WG 32
#define NWG_LAYER   32
#define NWG_TOTAL   33
#define THREADS     128

#define BETA_   0.96f
#define THRESH_ 1.0f

// d_out layout (flat, float): out_spk | out_mem | tr1 | tr2
#define OUT_MEM_OFF (B_*T_*NOUT_)            // 2,621,440
#define TR1_OFF     (2*B_*T_*NOUT_)          // 5,242,880
#define TR2_OFF     (TR1_OFF + B_*H1_)       // 5,275,648

__device__ inline v8f vzero() {
  v8f z;
#pragma unroll
  for (int i = 0; i < 8; ++i) z[i] = 0.0f;
  return z;
}

__device__ inline v8f wmma_f16(v16h a, v16h b, v8f c) {
  // D = A(16x32 f16) * B(32x16 f16) + C(16x16 f32)
  return __builtin_amdgcn_wmma_f32_16x16x32_f16(
      /*neg_a=*/false, a, /*neg_b=*/false, b,
      /*c_mod=*/(short)0, c, /*reuse_a=*/false, /*reuse_b=*/false);
}

// Load a 16-half WMMA operand fragment for one lane from a row-major f16
// matrix row slice. p0 = row_base + kchunk_base + (lane>=16 ? 8 : 0).
// ISA layout: halves 0..7 -> K = base+0..7 ; halves 8..15 -> K = base+16..23.
__device__ inline v16h load_frag(const _Float16* p0) {
  union { v16h h; uint32_t u[8]; } f;
  const uint32_t* q = (const uint32_t*)p0;
  f.u[0] = q[0]; f.u[1] = q[1]; f.u[2] = q[2];  f.u[3] = q[3];
  f.u[4] = q[8]; f.u[5] = q[9]; f.u[6] = q[10]; f.u[7] = q[11];
  return f.h;
}

// snntorch Leaky (reset_mechanism='subtract'): reset from incoming mem,
// mem = beta*mem + cur - reset*thresh, spike from new mem.
__device__ inline void lif_step(v8f& mem, const v8f& cur, v8f& spk) {
#pragma unroll
  for (int i = 0; i < 8; ++i) {
    float r = (mem[i] > THRESH_) ? 1.0f : 0.0f;
    float m = BETA_ * mem[i] + cur[i] - r * THRESH_;
    mem[i] = m;
    spk[i] = (m > THRESH_) ? 1.0f : 0.0f;
  }
}

// CDNA5 async global->LDS copy (ASYNCcnt-tracked), 16 bytes per lane.
__device__ inline void async_copy_b128(uint32_t lds_off, const void* gaddr) {
  asm volatile("global_load_async_to_lds_b128 %0, %1, off"
               :: "v"(lds_off), "v"(gaddr)
               : "memory");
}
__device__ inline void wait_async0() {
  asm volatile("s_wait_asynccnt 0" ::: "memory");
}

// Sense-reversing grid barrier over `members` blocks (persistent kernel,
// all blocks resident). Independent (cnt,gen) pairs per barrier instance.
__device__ inline void grid_barrier(unsigned* cnt, unsigned* gen, unsigned members) {
  __syncthreads();
  if (threadIdx.x == 0) {
    __threadfence();
    unsigned g = __hip_atomic_load(gen, __ATOMIC_RELAXED, __HIP_MEMORY_SCOPE_AGENT);
    unsigned arrived =
        __hip_atomic_fetch_add(cnt, 1u, __ATOMIC_ACQ_REL, __HIP_MEMORY_SCOPE_AGENT);
    if (arrived == members - 1) {
      __hip_atomic_store(cnt, 0u, __ATOMIC_RELAXED, __HIP_MEMORY_SCOPE_AGENT);
      __hip_atomic_fetch_add(gen, 1u, __ATOMIC_ACQ_REL, __HIP_MEMORY_SCOPE_AGENT);
    } else {
      while (__hip_atomic_load(gen, __ATOMIC_ACQUIRE, __HIP_MEMORY_SCOPE_AGENT) == g)
        __builtin_amdgcn_s_sleep(1);
    }
    __threadfence();
  }
  __syncthreads();
}

__global__ void snn_init_kernel(unsigned* bar) {
  if (threadIdx.x < 4) bar[threadIdx.x] = 0u;   // P1: cnt,gen ; P2: cnt,gen
}

__global__ void __launch_bounds__(THREADS)
snn_persistent_kernel(const float* __restrict__ inp,
                      const float* __restrict__ W1,
                      const float* __restrict__ W2,
                      const float* __restrict__ Wout,
                      float* __restrict__ out,
                      unsigned* __restrict__ bar,
                      _Float16* __restrict__ gspk1,
                      _Float16* __restrict__ gspk2) {
  extern __shared__ _Float16 smem[];
  const int wg   = blockIdx.x;
  const int tid  = threadIdx.x;
  const int lane = tid & 31;
  const int wave = tid >> 5;        // 4 waves/WG
  const int mt   = wave & 1;        // M-tile (batch 0-15 / 16-31)
  const int nt   = wave >> 1;       // N-tile within WG's 32 columns
  const int lm   = lane & 15;
  const int hi   = (lane < 16) ? 0 : 8;

  unsigned* p1 = bar;      // layer WGs only (32 members)
  unsigned* p2 = bar + 2;  // all WGs (33 members)

  if (wg < NWG_LAYER) {
    // ---------------- layer workgroups: own 32 cols of H1 and 32 cols of H2
    _Float16* sW1 = smem;                          // 32 x K1S
    _Float16* sW2 = sW1 + COLS_PER_WG * K1S;       // 32 x K2S
    _Float16* sX0 = sW2 + COLS_PER_WG * K2S;       // 32 x K1S  (x_t, even t)
    _Float16* sX1 = sX0 + B_ * K1S;                // 32 x K1S  (x_t, odd t)
    _Float16* sS1 = sX1 + B_ * K1S;                // 32 x K2S  (full spk1)
    const int colBase = wg * COLS_PER_WG;

    // one-time weight load: fp32 global -> f16 LDS (zero-padded K)
    for (int idx = tid; idx < COLS_PER_WG * K1S; idx += THREADS) {
      int c = idx / K1S, k = idx % K1S;
      float v = (k < NIN_) ? W1[(colBase + c) * NIN_ + k] : 0.0f;
      sW1[idx] = (_Float16)v;
    }
    for (int idx = tid; idx < COLS_PER_WG * K2S; idx += THREADS) {
      int c = idx / K2S, k = idx % K2S;
      float v = (k < H1_) ? W2[(colBase + c) * H1_ + k] : 0.0f;
      sW2[idx] = (_Float16)v;
    }
    // stage x_0 into buffer 0
    for (int idx = tid; idx < B_ * K1S; idx += THREADS) {
      int b = idx / K1S, n = idx % K1S;
      float v = (n < NIN_) ? inp[b * (T_ * NIN_) + n] : 0.0f;
      sX0[idx] = (_Float16)v;
    }
    __syncthreads();

    // persistent per-wave register state: membranes + spike traces
    v8f mem1 = vzero(), tr1 = vzero(), mem2 = vzero(), tr2 = vzero();

    const int aoff1 = (mt * 16 + lm) * K1S + hi;
    const _Float16* bRow1 = sW1 + (nt * 16 + lm) * K1S + hi;
    const _Float16* aRow2 = sS1 + (mt * 16 + lm) * K2S + hi;
    const _Float16* bRow2 = sW2 + (nt * 16 + lm) * K2S + hi;
    const uint32_t  sS1off = (uint32_t)(uintptr_t)sS1;

    for (int t = 0; t < T_; ++t) {
      _Float16* sXcur  = (t & 1) ? sX1 : sX0;
      _Float16* sXnext = (t & 1) ? sX0 : sX1;

      // ---- layer 1 GEMM: cur1 = x_t @ W1[colBase..+32].T (dual accumulators)
      const _Float16* aRow1 = sXcur + aoff1;
      v8f acc0 = vzero(), acc1 = vzero();
      for (int kc = 0; kc < NK1; kc += 2) {
        acc0 = wmma_f16(load_frag(aRow1 + kc * 32),
                        load_frag(bRow1 + kc * 32), acc0);
        acc1 = wmma_f16(load_frag(aRow1 + kc * 32 + 32),
                        load_frag(bRow1 + kc * 32 + 32), acc1);
      }
      v8f spk;
      {
        v8f acc = acc0 + acc1;
        lif_step(mem1, acc, spk);
      }
      tr1 += spk;
      {
        _Float16* dst = gspk1 + (t & 1) * (B_ * H1_);
        int col = colBase + nt * 16 + lm;
#pragma unroll
        for (int r = 0; r < 8; ++r)
          dst[(mt * 16 + r + hi) * H1_ + col] = (_Float16)spk[r];
      }

      // prefetch x(t+1): overlaps with the barrier spin below
      if (t + 1 < T_) {
        for (int idx = tid; idx < B_ * K1S; idx += THREADS) {
          int b = idx / K1S, n = idx % K1S;
          float v = (n < NIN_) ? inp[b * (T_ * NIN_) + (t + 1) * NIN_ + n] : 0.0f;
          sXnext[idx] = (_Float16)v;
        }
      }

      grid_barrier(p1, p1 + 1, NWG_LAYER);   // spk1(t) complete, visible

      // stage full spk1 into LDS via async global->LDS (padded stride)
      {
        const _Float16* src = gspk1 + (t & 1) * (B_ * H1_);
        for (int c = tid; c < B_ * (H1_ / 8); c += THREADS) {
          int b = c >> 7, j = c & 127;   // 128 chunks of 8 halves per row
          async_copy_b128(sS1off + (uint32_t)((b * K2S + j * 8) * 2),
                          (const void*)(src + b * H1_ + j * 8));
        }
        wait_async0();
        __syncthreads();
      }

      // ---- layer 2 GEMM: cur2 = spk1 @ W2[colBase..+32].T
      acc0 = vzero(); acc1 = vzero();
      for (int kc = 0; kc < NK2; kc += 2) {
        acc0 = wmma_f16(load_frag(aRow2 + kc * 32),
                        load_frag(bRow2 + kc * 32), acc0);
        acc1 = wmma_f16(load_frag(aRow2 + kc * 32 + 32),
                        load_frag(bRow2 + kc * 32 + 32), acc1);
      }
      {
        v8f acc = acc0 + acc1;
        lif_step(mem2, acc, spk);
      }
      tr2 += spk;
      {
        _Float16* dst = gspk2 + (t & 1) * (B_ * H2_);
        int col = colBase + nt * 16 + lm;
#pragma unroll
        for (int r = 0; r < 8; ++r)
          dst[(mt * 16 + r + hi) * H2_ + col] = (_Float16)spk[r];
      }
      grid_barrier(p2, p2 + 1, NWG_TOTAL);   // spk2(t) complete, visible
    }

    // write spike traces
    {
      int col = colBase + nt * 16 + lm;
#pragma unroll
      for (int r = 0; r < 8; ++r) {
        int b = mt * 16 + r + hi;
        out[TR1_OFF + b * H1_ + col] = tr1[r];
        out[TR2_OFF + b * H2_ + col] = tr2[r];
      }
    }
  } else {
    // ---------------- readout workgroup: Wout (20 rows padded to 32)
    _Float16* sWo = smem;                     // 32 x K2S
    _Float16* sS2 = sWo + 32 * K2S;           // 32 x K2S (full spk2)
    for (int idx = tid; idx < 32 * K2S; idx += THREADS) {
      int c = idx / K2S, k = idx % K2S;
      float v = (c < NOUT_ && k < H2_) ? Wout[c * H2_ + k] : 0.0f;
      sWo[idx] = (_Float16)v;
    }
    __syncthreads();

    v8f memo = vzero();
    const _Float16* aRow = sS2 + (mt * 16 + lm) * K2S + hi;
    const _Float16* bRow = sWo + (nt * 16 + lm) * K2S + hi;
    const uint32_t  sS2off = (uint32_t)(uintptr_t)sS2;
    const int n = nt * 16 + lm;

    for (int t = 0; t < T_; ++t) {
      grid_barrier(p2, p2 + 1, NWG_TOTAL);   // spk2(t) ready
      {
        const _Float16* src = gspk2 + (t & 1) * (B_ * H2_);
        for (int c = tid; c < B_ * (H2_ / 8); c += THREADS) {
          int b = c >> 7, j = c & 127;
          async_copy_b128(sS2off + (uint32_t)((b * K2S + j * 8) * 2),
                          (const void*)(src + b * H2_ + j * 8));
        }
        wait_async0();
        __syncthreads();
      }
      v8f acc0 = vzero(), acc1 = vzero();
      for (int kc = 0; kc < NK2; kc += 2) {
        acc0 = wmma_f16(load_frag(aRow + kc * 32),
                        load_frag(bRow + kc * 32), acc0);
        acc1 = wmma_f16(load_frag(aRow + kc * 32 + 32),
                        load_frag(bRow + kc * 32 + 32), acc1);
      }
      v8f spk;
      {
        v8f acc = acc0 + acc1;
        lif_step(memo, acc, spk);
      }
      if (n < NOUT_) {
#pragma unroll
        for (int r = 0; r < 8; ++r) {
          int b = mt * 16 + r + hi;
          out[(b * T_ + t) * NOUT_ + n]               = spk[r];
          out[OUT_MEM_OFF + (b * T_ + t) * NOUT_ + n] = memo[r];
        }
      }
      __syncthreads();   // protect sS2 before next restage
    }
  }
}

extern "C" void kernel_launch(void* const* d_in, const int* in_sizes, int n_in,
                              void* d_out, int out_size, void* d_ws, size_t ws_size,
                              hipStream_t stream) {
  (void)in_sizes; (void)n_in; (void)out_size; (void)ws_size;
  const float* inp  = (const float*)d_in[0];
  // d_in[1] = target, d_in[2] = bf : unused in eval forward
  const float* W1   = (const float*)d_in[3];
  const float* W2   = (const float*)d_in[4];
  const float* Wout = (const float*)d_in[5];
  float* out = (float*)d_out;

  unsigned*  bar   = (unsigned*)d_ws;                         // 4 uints
  _Float16*  gspk1 = (_Float16*)((char*)d_ws + 256);          // 2 x 32 x 1024
  _Float16*  gspk2 = gspk1 + 2 * B_ * H1_;                    // 2 x 32 x 1024

  snn_init_kernel<<<1, 32, 0, stream>>>(bar);

  size_t lds_bytes =
      (size_t)(COLS_PER_WG * K1S + COLS_PER_WG * K2S + 2 * B_ * K1S + B_ * K2S) *
      sizeof(_Float16);  // 267,520 B < 320 KB/WGP
  snn_persistent_kernel<<<NWG_TOTAL, THREADS, lds_bytes, stream>>>(
      inp, W1, W2, Wout, out, bar, gspk1, gspk2);
}
